// ResonantColorMatcher_22393959481792
// MI455X (gfx1250) — compile-verified, matrix operand order checked by reading, not run on hardware
//
#include <hip/hip_runtime.h>

// ---------------------------------------------------------------------------
// ResonantColorMatcher forward for gfx1250 (MI455X, wave32, WMMA).
// Channel-mixing convs = v_wmma_f32_16x16x32_f16 GEMMs: f16 weights packed
// per call (contiguous A rows -> global_load_b128), activation tiles staged
// transposed in LDS (coalesced float4 loads -> b16 stores -> ds_load_b128
// fragment reads). All border/concat handling is branchless or K-chunk
// uniform; no per-element exec juggling in the WMMA path.
// ---------------------------------------------------------------------------

#define PLI 65536            // 256*256 pixels
#define IMG 256
#define DT_C 0.25f

typedef __attribute__((ext_vector_type(16))) _Float16 v16h;
typedef __attribute__((ext_vector_type(8)))  _Float16 v8h;
typedef __attribute__((ext_vector_type(8)))  float    v8f;

__device__ __forceinline__ float lrelu_f(float x) { return x > 0.f ? x : 0.2f * x; }
__device__ __forceinline__ float sigm_f(float x)  { return 1.f / (1.f + __expf(-x)); }
__device__ __forceinline__ float act_f(float x, int act) {
  if (act == 1) return lrelu_f(x);
  if (act == 2) return sigm_f(x);
  if (act == 3) return x * sigm_f(x);   // silu
  return x;
}
__device__ __forceinline__ int clampi(int v, int lo, int hi) {
  return v < lo ? lo : (v > hi ? hi : v);
}

// ---------------------------------------------------------------------------
// Pointwise GEMM: out[co,p] = act(sum_ci W[co,ci]*in[ci,p] + b[co])
// Wh f16, CoutP % 16 == 0, Cin % 32 == 0, Csplit % 32 == 0 (chunk-aligned).
// Block = 8 waves = 4 co-tiles x 2 px sub-tiles over one 32ci x 32px tile.
// LDS tile stored transposed [px][k] so B fragments are 2 x ds_load_b128.
// ---------------------------------------------------------------------------
template <int MODE2, int ACCUM>
__global__ void k_pw_wmma(const _Float16* __restrict__ Wh,
                          const float* __restrict__ in, const float* __restrict__ in2,
                          const float* __restrict__ bias, float* __restrict__ out,
                          int Cin, int Csplit, int CoutP, int act)
{
  __shared__ _Float16 sBT[32][40];                // [px][k], stride 80B (16B aligned)
  const int tid  = threadIdx.x;
  const int lane = tid & 31;
  const int wave = tid >> 5;
  const int p0   = blockIdx.x * 32;
  const int px0  = (wave & 1) * 16;
  const int co0  = (blockIdx.y * 4 + (wave >> 1)) * 16;
  const bool active = co0 < CoutP;                // wave-uniform
  const int mn    = lane & 15;
  const int khalf = (lane & 16) ? 8 : 0;
  const int ci_l  = tid >> 3;                     // staging: 0..31
  const int pxl   = (tid & 7) * 4;                // staging: 0,4,..,28

  v8f acc = {};
  for (int kc = 0; kc < Cin; kc += 32) {
    __syncthreads();                              // protect prior-iter reads
    {                                             // stage 32ci x 32px (transposed)
      float4 vv;
      if (MODE2 == 2 && kc >= Csplit) {           // uniform per chunk
        const float s = in2[kc + ci_l - Csplit];
        vv = make_float4(s, s, s, s);
      } else {
        const float* base = (MODE2 == 1 && kc >= Csplit)
                          ? in2 + (size_t)(kc + ci_l - Csplit) * PLI
                          : in  + (size_t)(kc + ci_l) * PLI;
        vv = *(const float4*)&base[p0 + pxl];
        if (kc + 32 < Cin)
          __builtin_prefetch(&in[(kc + 32 + ci_l) * PLI + p0 + pxl], 0, 1);
      }
      sBT[pxl + 0][ci_l] = (_Float16)vv.x;
      sBT[pxl + 1][ci_l] = (_Float16)vv.y;
      sBT[pxl + 2][ci_l] = (_Float16)vv.z;
      sBT[pxl + 3][ci_l] = (_Float16)vv.w;
    }
    __syncthreads();
    if (active) {
      const _Float16* wrow = Wh + (size_t)(co0 + mn) * Cin + kc;
      const v8h alo = *(const v8h*)(wrow + khalf);            // global_load_b128
      const v8h ahi = *(const v8h*)(wrow + 16 + khalf);       // global_load_b128
      const v8h blo = *(const v8h*)&sBT[px0 + mn][khalf];     // ds_load_b128
      const v8h bhi = *(const v8h*)&sBT[px0 + mn][16 + khalf];// ds_load_b128
      v16h a, b;
#pragma unroll
      for (int e = 0; e < 8; ++e) {
        a[e] = alo[e]; a[8 + e] = ahi[e];
        b[e] = blo[e]; b[8 + e] = bhi[e];
      }
      acc = __builtin_amdgcn_wmma_f32_16x16x32_f16(false, a, false, b, (short)0, acc,
                                                   false, false);
    }
  }
  if (active) {
    const int pix = p0 + px0 + mn;
#pragma unroll
    for (int r = 0; r < 8; ++r) {
      const int co = co0 + r + ((lane & 16) ? 8 : 0);
      float v = acc[r] + (bias ? bias[co] : 0.f);
      v = act_f(v, act);
      if (ACCUM) v += out[co * PLI + pix];
      out[co * PLI + pix] = v;
    }
  }
}

// ---------------------------------------------------------------------------
// 3x3 conv (pad=1), Cin=Cout=64, implicit GEMM: K = 9 taps x 64.
// Wc f16 repacked [tap][co][ci]. One 34-px halo row staged per (ky, K-chunk);
// the 3 kx taps reuse it with shifted ds_load_b128 rows (6 stages / block).
// Border clamp is branchless (clamped address + v_cndmask zero).
// ---------------------------------------------------------------------------
__global__ void k_c3_wmma(const _Float16* __restrict__ Wc, const float* __restrict__ in,
                          const float* __restrict__ bias, float* __restrict__ out, int act)
{
  __shared__ _Float16 sBT[34][40];                // rows: px -1..32 ; cols: k
  const int tid  = threadIdx.x;
  const int lane = tid & 31;
  const int wave = tid >> 5;
  const int p0   = blockIdx.x * 32;
  const int y    = p0 >> 8;
  const int x0   = p0 & 255;
  const int px0  = (wave & 1) * 16;
  const int co0  = (wave >> 1) * 16;
  const int mn    = lane & 15;
  const int khalf = (lane & 16) ? 8 : 0;

  v8f acc = {};
  for (int ky = 0; ky < 3; ++ky) {
    const int yy  = y + ky - 1;
    const bool yok = (yy >= 0) && (yy < IMG);
    const int yc  = clampi(yy, 0, IMG - 1);
    for (int kc = 0; kc < 64; kc += 32) {
      __syncthreads();
      {                                           // stage 34 rows x 32 ci
        const int rr  = tid & 31;                 // coalesced over x
        const int cig = tid >> 5;                 // 0..7
        const int xx  = x0 + rr - 1;
        const int xc  = clampi(xx, 0, IMG - 1);
        const bool ok = yok && (xx >= 0) && (xx < IMG);
#pragma unroll
        for (int k4 = 0; k4 < 4; ++k4) {
          const int cl = cig + 8 * k4;            // 0..31
          float v = in[(size_t)(kc + cl) * PLI + yc * IMG + xc];
          sBT[rr][cl] = (_Float16)(ok ? v : 0.f);
        }
        if (tid < 64) {                           // rows 32,33
          const int r2  = 32 + (tid >> 5);
          const int cl  = tid & 31;
          const int xe  = x0 + r2 - 1;
          const int xec = clampi(xe, 0, IMG - 1);
          const bool ok2 = yok && (xe >= 0) && (xe < IMG);
          float v = in[(size_t)(kc + cl) * PLI + yc * IMG + xec];
          sBT[r2][cl] = (_Float16)(ok2 ? v : 0.f);
        }
      }
      __syncthreads();
#pragma unroll
      for (int kx = 0; kx < 3; ++kx) {
        const int tap = ky * 3 + kx;
        const _Float16* wrow = Wc + tap * 4096 + (co0 + mn) * 64 + kc;
        const v8h alo = *(const v8h*)(wrow + khalf);
        const v8h ahi = *(const v8h*)(wrow + 16 + khalf);
        const int row = px0 + mn + kx;            // input col = x0+px+kx-1
        const v8h blo = *(const v8h*)&sBT[row][khalf];
        const v8h bhi = *(const v8h*)&sBT[row][16 + khalf];
        v16h a, b;
#pragma unroll
        for (int e = 0; e < 8; ++e) {
          a[e] = alo[e]; a[8 + e] = ahi[e];
          b[e] = blo[e]; b[8 + e] = bhi[e];
        }
        acc = __builtin_amdgcn_wmma_f32_16x16x32_f16(false, a, false, b, (short)0, acc,
                                                     false, false);
      }
    }
  }
  const int pix = p0 + px0 + mn;
#pragma unroll
  for (int r = 0; r < 8; ++r) {
    const int co = co0 + r + ((lane & 16) ? 8 : 0);
    out[co * PLI + pix] = act_f(acc[r] + bias[co], act);
  }
}

// ------------------------- weight prep (per call) ---------------------------
__global__ void k_cvt(const float* __restrict__ w, _Float16* __restrict__ o, int n) {
  const int i = blockIdx.x * blockDim.x + threadIdx.x;
  if (i < n) o[i] = (_Float16)w[i];
}
__global__ void k_cvt_pad(const float* __restrict__ w, _Float16* __restrict__ o,
                          int Cout, int CoutP, int Cin) {
  const int i = blockIdx.x * blockDim.x + threadIdx.x;
  if (i >= CoutP * Cin) return;
  const int co = i / Cin, ci = i - co * Cin;
  o[i] = (co < Cout) ? (_Float16)w[co * Cin + ci] : (_Float16)0.f;
}
__global__ void k_c3repack(const float* __restrict__ w, _Float16* __restrict__ o) {
  const int i = blockIdx.x * blockDim.x + threadIdx.x;   // 9*64*64
  if (i >= 36864) return;
  const int tap = i >> 12, co = (i >> 6) & 63, ci = i & 63;
  o[i] = (_Float16)w[(co * 64 + ci) * 9 + tap];
}
__global__ void k_bias_pad(const float* __restrict__ b, float* __restrict__ o, int Cout) {
  const int i = threadIdx.x;
  if (i < 16) o[i] = (i < Cout) ? b[i] : 0.f;
}

// ---------------------------------------------------------------------------
// Depthwise 3x3 (dilation dil, pad=dil) -- branchless border handling.
// ---------------------------------------------------------------------------
__global__ void k_dw3x3(const float* __restrict__ in, const float* __restrict__ W,
                        const float* __restrict__ bias, float* __restrict__ out,
                        int dil, int act)
{
  const int idx = blockIdx.x * blockDim.x + threadIdx.x;
  if (idx >= 64 * PLI) return;
  const int c = idx >> 16;
  const int p = idx & (PLI - 1);
  const int y = p >> 8, x = p & 255;
  float s = bias ? bias[c] : 0.f;
#pragma unroll
  for (int ky = 0; ky < 3; ++ky)
#pragma unroll
    for (int kx = 0; kx < 3; ++kx) {
      const int yy = y + (ky - 1) * dil;
      const int xx = x + (kx - 1) * dil;
      const bool ok = (yy >= 0) && (yy < IMG) && (xx >= 0) && (xx < IMG);
      const float v = in[c * PLI + clampi(yy, 0, IMG - 1) * IMG + clampi(xx, 0, IMG - 1)];
      s += (ok ? v : 0.f) * W[c * 9 + ky * 3 + kx];
    }
  out[idx] = act_f(s, act);
}

// 3-input-channel 3x3 conv (sb_w0 / re_w0) -- branchless borders.
__global__ void k_conv3x3_c3(const float* __restrict__ in, const float* __restrict__ W,
                             const float* __restrict__ bias, float* __restrict__ out,
                             int act)
{
  const int idx = blockIdx.x * blockDim.x + threadIdx.x;
  if (idx >= 64 * PLI) return;
  const int co = idx >> 16;
  const int p = idx & (PLI - 1);
  const int y = p >> 8, x = p & 255;
  float s = bias[co];
#pragma unroll
  for (int tap = 0; tap < 9; ++tap) {
    const int yy = y + tap / 3 - 1;
    const int xx = x + tap % 3 - 1;
    const bool ok = (yy >= 0) && (yy < IMG) && (xx >= 0) && (xx < IMG);
    const int off = clampi(yy, 0, IMG - 1) * IMG + clampi(xx, 0, IMG - 1);
    for (int ci = 0; ci < 3; ++ci) {
      const float v = in[ci * PLI + off];
      s += (ok ? v : 0.f) * W[(co * 3 + ci) * 9 + tap];
    }
  }
  out[idx] = act_f(s, act);
}

// ------------------------- elementwise helpers ------------------------------
__global__ void k_gray(const float* __restrict__ src, float* __restrict__ g) {
  const int p = blockIdx.x * blockDim.x + threadIdx.x;
  if (p < PLI) g[p] = (src[p] + src[PLI + p] + src[2 * PLI + p]) * (1.f / 3.f);
}
__global__ void k_il0(const float* __restrict__ src, const float* __restrict__ gray,
                      const float* __restrict__ w, float* __restrict__ out) {
  const int idx = blockIdx.x * blockDim.x + threadIdx.x;
  if (idx >= 64 * PLI) return;
  const int co = idx >> 16, p = idx & (PLI - 1);
  float s = w[co * 4 + 0] * src[p] + w[co * 4 + 1] * src[PLI + p] +
            w[co * 4 + 2] * src[2 * PLI + p] + w[co * 4 + 3] * gray[p];
  out[idx] = lrelu_f(s);
}
__global__ void k_sc(const float* __restrict__ f, const float* __restrict__ src,
                     const float* __restrict__ w, const float* __restrict__ b,
                     float* __restrict__ out) {
  const int idx = blockIdx.x * blockDim.x + threadIdx.x;
  if (idx >= 64 * PLI) return;
  const int co = idx >> 16, p = idx & (PLI - 1);
  out[idx] = f[idx] + w[co * 3 + 0] * src[p] + w[co * 3 + 1] * src[PLI + p] +
             w[co * 3 + 2] * src[2 * PLI + p] + b[co];
}
__global__ void k_drive(const float* __restrict__ ilf, const float* __restrict__ imap,
                        float* __restrict__ d) {
  const int idx = blockIdx.x * blockDim.x + threadIdx.x;
  if (idx < 64 * PLI) d[idx] = ilf[idx] * imap[idx & (PLI - 1)];
}
__global__ void k_axpb(const float* __restrict__ a, const float* __restrict__ b,
                       float alpha, float* __restrict__ o, int n) {
  const int i = blockIdx.x * blockDim.x + threadIdx.x;
  if (i < n) o[i] = a[i] + alpha * b[i];
}
__global__ void k_copy(const float* __restrict__ s, float* __restrict__ d, int n) {
  const int i = blockIdx.x * blockDim.x + threadIdx.x;
  if (i < n) d[i] = s[i];
}
__global__ void k_fill0(float* __restrict__ p, int n) {
  const int i = blockIdx.x * blockDim.x + threadIdx.x;
  if (i < n) p[i] = 0.f;
}
// In-place wm,zm,gm,bm -> damp, omega^2, gamma, beta (256-plane buffer).
__global__ void k_wzgb(float* __restrict__ w) {
  const int idx = blockIdx.x * blockDim.x + threadIdx.x;
  if (idx >= 64 * PLI) return;
  const int c = idx >> 16, p = idx & (PLI - 1);
  const float wm = w[c * PLI + p];
  const float zm = w[(64 + c) * PLI + p];
  const float gm = w[(128 + c) * PLI + p];
  const float bm = w[(192 + c) * PLI + p];
  const float omega = 2.f * sigm_f(wm);
  const float zeta  = sigm_f(zm);
  w[c * PLI + p]         = 2.f * zeta * omega;   // damp
  w[(64 + c) * PLI + p]  = omega * omega;        // om2
  w[(128 + c) * PLI + p] = 2.f * sigm_f(gm);     // gamma
  w[(192 + c) * PLI + p] = tanhf(bm);            // beta
}
__global__ void k_osc(float* __restrict__ x, float* __restrict__ v,
                      const float* __restrict__ force, const float* __restrict__ cpl,
                      const float* __restrict__ wz) {
  const int idx = blockIdx.x * blockDim.x + threadIdx.x;
  if (idx >= 64 * PLI) return;
  const int c = idx >> 16, p = idx & (PLI - 1);
  const float damp = wz[c * PLI + p];
  const float om2  = wz[(64 + c) * PLI + p];
  const float a = force[idx] + cpl[idx] - damp * v[idx] - om2 * x[idx];
  const float vv = v[idx] + a * DT_C;
  x[idx] = x[idx] + vv * DT_C;
  v[idx] = vv;
}
__global__ void k_inorm_apply(const float* __restrict__ force, const float* __restrict__ x,
                              const float* __restrict__ wz, const float* __restrict__ mean,
                              const float* __restrict__ rstd, float* __restrict__ out) {
  const int idx = blockIdx.x * blockDim.x + threadIdx.x;
  if (idx >= 64 * PLI) return;
  const int c = idx >> 16, p = idx & (PLI - 1);
  const float val = (force[idx] + x[idx] - mean[c]) * rstd[c];
  out[idx] = wz[(128 + c) * PLI + p] * val + wz[(192 + c) * PLI + p];
}
__global__ void k_lsadd(const float* __restrict__ ob, const float* __restrict__ rr,
                        const float* __restrict__ ls, float* __restrict__ feat) {
  const int idx = blockIdx.x * blockDim.x + threadIdx.x;
  if (idx >= 64 * PLI) return;
  feat[idx] = ob[idx] + ls[idx >> 16] * rr[idx];
}
__global__ void k_mhead(const float* __restrict__ pm, const float* __restrict__ src,
                        float* __restrict__ expl) {
  const int p = blockIdx.x * blockDim.x + threadIdx.x;
  if (p >= PLI) return;
  float m[9];
#pragma unroll
  for (int j = 0; j < 9; ++j) m[j] = tanhf(pm[j * PLI + p]);
  const float s0 = src[p], s1 = src[PLI + p], s2 = src[2 * PLI + p];
#pragma unroll
  for (int o = 0; o < 3; ++o) {
    float acc = pm[(9 + o) * PLI + p];
    acc += ((o == 0 ? 1.f : 0.f) + 0.1f * m[o * 3 + 0]) * s0;
    acc += ((o == 1 ? 1.f : 0.f) + 0.1f * m[o * 3 + 1]) * s1;
    acc += ((o == 2 ? 1.f : 0.f) + 0.1f * m[o * 3 + 2]) * s2;
    expl[o * PLI + p] = sigm_f(acc);
  }
}
__global__ void k_final(const float* __restrict__ g, const float* __restrict__ spl,
                        const float* __restrict__ expl, float* __restrict__ out) {
  const int p = blockIdx.x * blockDim.x + threadIdx.x;
  if (p >= PLI) return;
  const float gg = g[p];
#pragma unroll
  for (int o = 0; o < 3; ++o)
    out[o * PLI + p] = gg * spl[o * PLI + p] + (1.f - gg) * expl[o * PLI + p];
}

// ------------------------- reductions (LDS) --------------------------------
__global__ void k_reduce_meanstd(const float* __restrict__ a, const float* __restrict__ b,
                                 float* __restrict__ mean, float* __restrict__ rstd) {
  __shared__ float ss[256], sq[256];
  const int c = blockIdx.x;
  float s = 0.f, q = 0.f;
  for (int i = threadIdx.x; i < PLI; i += 256) {
    const float v = a[c * PLI + i] + b[c * PLI + i];
    s += v; q += v * v;
  }
  ss[threadIdx.x] = s; sq[threadIdx.x] = q;
  __syncthreads();
  for (int o = 128; o > 0; o >>= 1) {
    if (threadIdx.x < o) { ss[threadIdx.x] += ss[threadIdx.x + o];
                           sq[threadIdx.x] += sq[threadIdx.x + o]; }
    __syncthreads();
  }
  if (threadIdx.x == 0) {
    const float m = ss[0] / (float)PLI;
    mean[c] = m;
    rstd[c] = rsqrtf(sq[0] / (float)PLI - m * m + 1e-5f);
  }
}
__global__ void k_stats_block(const float* __restrict__ r, float* __restrict__ sum4,
                              float* __restrict__ sumsq4) {
  __shared__ float ss[256], sq[256];
  const int c = blockIdx.x >> 4;
  const int blk = blockIdx.x & 15;
  const int by = blk >> 2, bx = blk & 3;
  float s = 0.f, q = 0.f;
  for (int i = threadIdx.x; i < 4096; i += 256) {
    const int ly = i >> 6, lx = i & 63;
    const float v = r[c * PLI + (by * 64 + ly) * IMG + (bx * 64 + lx)];
    s += v; q += v * v;
  }
  ss[threadIdx.x] = s; sq[threadIdx.x] = q;
  __syncthreads();
  for (int o = 128; o > 0; o >>= 1) {
    if (threadIdx.x < o) { ss[threadIdx.x] += ss[threadIdx.x + o];
                           sq[threadIdx.x] += sq[threadIdx.x + o]; }
    __syncthreads();
  }
  if (threadIdx.x == 0) { sum4[blockIdx.x] = ss[0]; sumsq4[blockIdx.x] = sq[0]; }
}
__global__ void k_stats_z(const float* __restrict__ sum4, const float* __restrict__ sumsq4,
                          float* __restrict__ z) {
  const int c = threadIdx.x;
  if (c >= 64) return;
  float s1 = 0.f, q1 = 0.f;
  float s2[4] = {0.f, 0.f, 0.f, 0.f}, q2[4] = {0.f, 0.f, 0.f, 0.f};
  for (int b = 0; b < 16; ++b) {
    const float sb = sum4[c * 16 + b], qb = sumsq4[c * 16 + b];
    s1 += sb; q1 += qb;
    const int by = b >> 2, bx = b & 3;
    const int qi = (by >> 1) * 2 + (bx >> 1);
    s2[qi] += sb; q2[qi] += qb;
    const float mu = sb / 4096.f;
    z[640 + c * 16 + b]  = mu;
    z[1664 + c * 16 + b] = sqrtf(fmaxf(qb / 4096.f - mu * mu, 1e-6f));
  }
  const float mu1 = s1 / 65536.f;
  z[c] = mu1;
  z[64 + c] = sqrtf(fmaxf(q1 / 65536.f - mu1 * mu1, 1e-6f));
  for (int q = 0; q < 4; ++q) {
    const float mu2 = s2[q] / 16384.f;
    z[128 + c * 4 + q] = mu2;
    z[384 + c * 4 + q] = sqrtf(fmaxf(q2[q] / 16384.f - mu2 * mu2, 1e-6f));
  }
}
__global__ void k_fc(const float* __restrict__ in, const float* __restrict__ W,
                     const float* __restrict__ bias, float* __restrict__ out,
                     int nin, int nout, int act) {
  const int o = blockIdx.x * blockDim.x + threadIdx.x;
  if (o >= nout) return;
  float s = bias[o];
  for (int i = 0; i < nin; ++i) s += W[o * nin + i] * in[i];
  out[o] = act_f(s, act);
}

// ---------------------------------------------------------------------------
// Host side
// ---------------------------------------------------------------------------
enum {
  I_SRC = 0, I_REF = 1,
  I_L_c_dw, I_L_c_pw, I_L_hy_b0, I_L_hy_b1, I_L_hy_b2, I_L_hy_w0, I_L_hy_w1, I_L_hy_w2,
  I_L_ls, I_L_rf_b2, I_L_rf_b_dw, I_L_rf_b_pw, I_L_rf_dw, I_L_rf_pw, I_L_rf_w2,
  I_fu_b, I_fu_w, I_gt_b0, I_gt_b1, I_gt_w0, I_gt_w1,
  I_il_dw, I_il_head_b, I_il_head_w, I_il_pw, I_il_w0,
  I_mh_b0, I_mh_b1, I_mh_w0, I_mh_w1,
  I_pr_b0, I_pr_b1, I_pr_w0, I_pr_w1,
  I_re_b0, I_re_b1, I_re_w0, I_re_w1,
  I_sb_b0, I_sb_b1, I_sb_b2, I_sb_w0, I_sb_w1, I_sb_w2,
  I_sc_b, I_sc_w,
  I_sp_b0, I_sp_b1, I_sp_b2, I_sp_w0, I_sp_w1, I_sp_w2
};

static constexpr size_t PLS    = 65536;
static constexpr size_t F_GRAY = 0;
static constexpr size_t F_IMAP = F_GRAY + PLS;          // 16 planes (padded)
static constexpr size_t F_PM   = F_IMAP + 16 * PLS;     // 16 planes (padded, 12 used)
static constexpr size_t F_G    = F_PM + 16 * PLS;       // 16 planes (padded)
static constexpr size_t F_SPL  = F_G + 16 * PLS;        // 16 planes (padded, 3 used)
static constexpr size_t F_EXPL = F_SPL + 16 * PLS;      // 3 planes
static constexpr size_t F_SM   = F_EXPL + 3 * PLS;      // 16K floats small scratch
static constexpr size_t F_WH   = F_SM + 16384;          // 256K floats = 512K f16 weights
static constexpr size_t F_A    = F_WH + 262144;         // 64-plane buffers...
static constexpr size_t F_B    = F_A + 64 * PLS;
static constexpr size_t F_ILF  = F_B + 64 * PLS;
static constexpr size_t F_FEAT = F_ILF + 64 * PLS;
static constexpr size_t F_DRV  = F_FEAT + 64 * PLS;
static constexpr size_t F_FRC  = F_DRV + 64 * PLS;
static constexpr size_t F_X    = F_FRC + 64 * PLS;
static constexpr size_t F_V    = F_X + 64 * PLS;
static constexpr size_t F_TMP  = F_V + 64 * PLS;        // 64 (also hy 32+32)
static constexpr size_t F_WZ   = F_TMP + 64 * PLS;      // 256 (also sp 128+128)
// small scratch sub-offsets (floats)
static constexpr size_t S_SUM4 = F_SM;
static constexpr size_t S_SSQ4 = F_SM + 1024;
static constexpr size_t S_Z    = F_SM + 2048;
static constexpr size_t S_ZH   = F_SM + 4736;
static constexpr size_t S_STY  = F_SM + 4864;
static constexpr size_t S_MEAN = F_SM + 4928;
static constexpr size_t S_RSTD = F_SM + 4992;
static constexpr size_t S_BIL  = F_SM + 5056;  // padded biases (16 each)
static constexpr size_t S_BMH  = F_SM + 5072;
static constexpr size_t S_BSP  = F_SM + 5088;
static constexpr size_t S_BGT  = F_SM + 5104;
// f16 weight sub-offsets (halves, within F_WH)
static constexpr size_t HO_sb_w1 = 0;
static constexpr size_t HO_sb_w2 = 36864;
static constexpr size_t HO_re_w1 = 73728;
static constexpr size_t HO_mh_w0 = 110592;
static constexpr size_t HO_il_pw = 147456;
static constexpr size_t HO_il_hd = 151552;
static constexpr size_t HO_fu    = 152576;
static constexpr size_t HO_hy_w0 = 160768;
static constexpr size_t HO_hy_w1 = 185344;
static constexpr size_t HO_hy_w2 = 191488;
static constexpr size_t HO_c_pw  = 240640;
static constexpr size_t HO_rf_pw = 265216;
static constexpr size_t HO_rf_w2 = 289792;
static constexpr size_t HO_mh_w1 = 314368;
static constexpr size_t HO_sp_w0 = 315392;
static constexpr size_t HO_sp_w1 = 323584;
static constexpr size_t HO_sp_w2 = 339968;
static constexpr size_t HO_gt_w0 = 342016;
static constexpr size_t HO_gt_w1 = 344064;

extern "C" void kernel_launch(void* const* d_in, const int* in_sizes, int n_in,
                              void* d_out, int out_size, void* d_ws, size_t ws_size,
                              hipStream_t stream)
{
  (void)in_sizes; (void)n_in; (void)out_size; (void)ws_size;
  const float* src = (const float*)d_in[I_SRC];
  const float* ref = (const float*)d_in[I_REF];
  float* ws  = (float*)d_ws;
  float* out = (float*)d_out;
  _Float16* WH = (_Float16*)(ws + F_WH);
  auto P = [&](int i) -> const float* { return (const float*)d_in[i]; };

  const dim3 B1(256);
  const dim3 G1(PLS / 256);
  const dim3 G64(64 * PLS / 256);

  auto cvt = [&](int pi, size_t ho, int n) {
    k_cvt<<<dim3((n + 255) / 256), B1, 0, stream>>>(P(pi), WH + ho, n);
  };
  auto cvtpad = [&](int pi, size_t ho, int Cout, int CoutP, int Cin) {
    k_cvt_pad<<<dim3((CoutP * Cin + 255) / 256), B1, 0, stream>>>(P(pi), WH + ho, Cout, CoutP, Cin);
  };
  auto pw = [&](int mode2, int accum, size_t ho, const float* in, const float* in2,
                const float* bias, float* o, int Cin, int Csplit, int CoutP, int act) {
    dim3 g(2048, (unsigned)((CoutP + 63) / 64));
    if (mode2 == 0 && accum == 0)
      k_pw_wmma<0, 0><<<g, B1, 0, stream>>>(WH + ho, in, in2, bias, o, Cin, Csplit, CoutP, act);
    else if (mode2 == 0)
      k_pw_wmma<0, 1><<<g, B1, 0, stream>>>(WH + ho, in, in2, bias, o, Cin, Csplit, CoutP, act);
    else if (mode2 == 1)
      k_pw_wmma<1, 0><<<g, B1, 0, stream>>>(WH + ho, in, in2, bias, o, Cin, Csplit, CoutP, act);
    else
      k_pw_wmma<2, 0><<<g, B1, 0, stream>>>(WH + ho, in, in2, bias, o, Cin, Csplit, CoutP, act);
  };
  auto c3w = [&](size_t ho, const float* in, const float* bias, float* o, int act) {
    k_c3_wmma<<<dim3(2048, 1), B1, 0, stream>>>(WH + ho, in, bias, o, act);
  };

  // ---- weight prep: f16 packing / padding (every call; deterministic) ----
  k_c3repack<<<dim3(144), B1, 0, stream>>>(P(I_sb_w1), WH + HO_sb_w1);
  k_c3repack<<<dim3(144), B1, 0, stream>>>(P(I_sb_w2), WH + HO_sb_w2);
  k_c3repack<<<dim3(144), B1, 0, stream>>>(P(I_re_w1), WH + HO_re_w1);
  k_c3repack<<<dim3(144), B1, 0, stream>>>(P(I_mh_w0), WH + HO_mh_w0);
  cvt(I_il_pw, HO_il_pw, 4096);
  cvt(I_fu_w,  HO_fu,    8192);
  cvt(I_L_hy_w0, HO_hy_w0, 24576);
  cvt(I_L_hy_w1, HO_hy_w1, 6144);
  cvt(I_L_hy_w2, HO_hy_w2, 49152);
  cvt(I_L_c_pw,  HO_c_pw,  24576);
  cvt(I_L_rf_pw, HO_rf_pw, 24576);
  cvt(I_L_rf_w2, HO_rf_w2, 24576);
  cvt(I_sp_w0, HO_sp_w0, 8192);
  cvt(I_sp_w1, HO_sp_w1, 16384);
  cvt(I_gt_w0, HO_gt_w0, 2048);
  cvtpad(I_il_head_w, HO_il_hd, 1, 16, 64);
  cvtpad(I_mh_w1,     HO_mh_w1, 12, 16, 64);
  cvtpad(I_sp_w2,     HO_sp_w2, 3, 16, 128);
  cvtpad(I_gt_w1,     HO_gt_w1, 1, 16, 32);
  k_bias_pad<<<dim3(1), dim3(16), 0, stream>>>(P(I_il_head_b), ws + S_BIL, 1);
  k_bias_pad<<<dim3(1), dim3(16), 0, stream>>>(P(I_mh_b1), ws + S_BMH, 12);
  k_bias_pad<<<dim3(1), dim3(16), 0, stream>>>(P(I_sp_b2), ws + S_BSP, 3);
  k_bias_pad<<<dim3(1), dim3(16), 0, stream>>>(P(I_gt_b1), ws + S_BGT, 1);

  // ---- illumination branch ----
  k_gray<<<G1, B1, 0, stream>>>(src, ws + F_GRAY);
  k_il0<<<G64, B1, 0, stream>>>(src, ws + F_GRAY, P(I_il_w0), ws + F_A);
  k_dw3x3<<<G64, B1, 0, stream>>>(ws + F_A, P(I_il_dw), nullptr, ws + F_B, 2, 0);
  pw(0, 0, HO_il_pw, ws + F_B, nullptr, nullptr, ws + F_ILF, 64, 64, 64, 1);
  pw(0, 0, HO_il_hd, ws + F_ILF, nullptr, ws + S_BIL, ws + F_IMAP, 64, 64, 16, 2);

  // ---- source backbone ----
  k_conv3x3_c3<<<G64, B1, 0, stream>>>(src, P(I_sb_w0), P(I_sb_b0), ws + F_A, 1);
  c3w(HO_sb_w1, ws + F_A, P(I_sb_b1), ws + F_B, 1);
  c3w(HO_sb_w2, ws + F_B, P(I_sb_b2), ws + F_A, 1);
  k_sc<<<G64, B1, 0, stream>>>(ws + F_A, src, P(I_sc_w), P(I_sc_b), ws + F_FEAT);
  pw(1, 0, HO_fu, ws + F_FEAT, ws + F_ILF, P(I_fu_b), ws + F_A, 128, 64, 64, 1);
  k_copy<<<G64, B1, 0, stream>>>(ws + F_A, ws + F_FEAT, 64 * PLS);

  // ---- reference statistics -> style ----
  k_conv3x3_c3<<<G64, B1, 0, stream>>>(ref, P(I_re_w0), P(I_re_b0), ws + F_A, 1);
  c3w(HO_re_w1, ws + F_A, P(I_re_b1), ws + F_B, 1);
  k_stats_block<<<dim3(1024), B1, 0, stream>>>(ws + F_B, ws + S_SUM4, ws + S_SSQ4);
  k_stats_z<<<dim3(1), dim3(64), 0, stream>>>(ws + S_SUM4, ws + S_SSQ4, ws + S_Z);
  k_fc<<<dim3(1), dim3(128), 0, stream>>>(ws + S_Z, P(I_pr_w0), P(I_pr_b0), ws + S_ZH, 2688, 128, 1);
  k_fc<<<dim3(1), dim3(64), 0, stream>>>(ws + S_ZH, P(I_pr_w1), P(I_pr_b1), ws + S_STY, 128, 64, 0);

  // ---- drive + oscillator state init ----
  k_drive<<<G64, B1, 0, stream>>>(ws + F_ILF, ws + F_IMAP, ws + F_DRV);
  k_fill0<<<G64, B1, 0, stream>>>(ws + F_X, 64 * PLS);
  k_fill0<<<G64, B1, 0, stream>>>(ws + F_V, 64 * PLS);

  // ---- resonant layers ----
  for (int l = 0; l < 6; ++l) {
    k_axpb<<<G64, B1, 0, stream>>>(ws + F_FEAT, ws + F_DRV, 0.2f, ws + F_FRC, 64 * PLS);
    pw(2, 0, HO_hy_w0 + l * 4096, ws + F_FRC, ws + S_STY, P(I_L_hy_b0) + l * 32,
       ws + F_TMP, 128, 64, 32, 1);
    pw(0, 0, HO_hy_w1 + l * 1024, ws + F_TMP, nullptr, P(I_L_hy_b1) + l * 32,
       ws + F_TMP + 32 * PLS, 32, 32, 32, 1);
    pw(0, 0, HO_hy_w2 + l * 8192, ws + F_TMP + 32 * PLS, nullptr, P(I_L_hy_b2) + l * 256,
       ws + F_WZ, 32, 32, 256, 0);
    k_wzgb<<<G64, B1, 0, stream>>>(ws + F_WZ);
    for (int s = 0; s < 10; ++s) {
      k_dw3x3<<<G64, B1, 0, stream>>>(ws + F_X, P(I_L_c_dw) + l * 576, nullptr,
                                      ws + F_TMP, 1, 0);
      pw(0, 1, HO_c_pw + l * 4096, ws + F_X, nullptr, nullptr, ws + F_TMP, 64, 64, 64, 0);
      k_osc<<<G64, B1, 0, stream>>>(ws + F_X, ws + F_V, ws + F_FRC, ws + F_TMP, ws + F_WZ);
    }
    k_reduce_meanstd<<<dim3(64), B1, 0, stream>>>(ws + F_FRC, ws + F_X,
                                                  ws + S_MEAN, ws + S_RSTD);
    k_inorm_apply<<<G64, B1, 0, stream>>>(ws + F_FRC, ws + F_X, ws + F_WZ,
                                          ws + S_MEAN, ws + S_RSTD, ws + F_B);
    k_dw3x3<<<G64, B1, 0, stream>>>(ws + F_B, P(I_L_rf_dw) + l * 576,
                                    P(I_L_rf_b_dw) + l * 64, ws + F_TMP, 1, 0);
    pw(0, 0, HO_rf_pw + l * 4096, ws + F_TMP, nullptr, P(I_L_rf_b_pw) + l * 64,
       ws + F_A, 64, 64, 64, 1);
    pw(0, 0, HO_rf_w2 + l * 4096, ws + F_A, nullptr, P(I_L_rf_b2) + l * 64,
       ws + F_TMP, 64, 64, 64, 0);
    k_lsadd<<<G64, B1, 0, stream>>>(ws + F_B, ws + F_TMP, P(I_L_ls) + l * 64, ws + F_FEAT);
  }

  // ---- matrix head (explicit path) ----
  c3w(HO_mh_w0, ws + F_FEAT, P(I_mh_b0), ws + F_A, 1);
  pw(0, 0, HO_mh_w1, ws + F_A, nullptr, ws + S_BMH, ws + F_PM, 64, 64, 16, 0);
  k_mhead<<<G1, B1, 0, stream>>>(ws + F_PM, src, ws + F_EXPL);

  // ---- spline path ----
  pw(0, 0, HO_sp_w0, ws + F_FEAT, nullptr, P(I_sp_b0), ws + F_WZ, 64, 64, 128, 3);
  pw(0, 0, HO_sp_w1, ws + F_WZ, nullptr, P(I_sp_b1), ws + F_WZ + 128 * PLS, 128, 128, 128, 3);
  pw(0, 0, HO_sp_w2, ws + F_WZ + 128 * PLS, nullptr, ws + S_BSP, ws + F_SPL, 128, 128, 16, 2);

  // ---- gate + final mix ----
  pw(0, 0, HO_gt_w0, ws + F_FEAT, nullptr, P(I_gt_b0), ws + F_TMP, 64, 64, 32, 1);
  pw(0, 0, HO_gt_w1, ws + F_TMP, nullptr, ws + S_BGT, ws + F_G, 32, 32, 16, 2);
  k_final<<<G1, B1, 0, stream>>>(ws + F_G, ws + F_SPL, ws + F_EXPL, out);
}
